// Seesaw_Conv_17729624998022
// MI455X (gfx1250) — matrix operand order repaired and checked by reading, not compiled, not run on hardware
//
#include <hip/hip_runtime.h>
#include <hip/hip_bf16.h>

// ---------------------------------------------------------------------------
// Problem constants (from reference)
// ---------------------------------------------------------------------------
#define BS   4
#define C_   16
#define H_   192
#define W_   192
#define KE_  9          // 3x3 kernel elements
#define OC_  128        // (KE-1)*C offset channels
#define KR_  144        // reduction K = C*3*3
#define KP_  160        // K padded to 5 * 32 for wmma 16x16x32
#define HW_  (H_*W_)            // 36864
#define BHW_ (BS*HW_)           // 147456
#define KEBHW_ (KE_*BHW_)       // 1327104
#define MT_  64                 // pixels per workgroup (4 x 16-pixel M tiles)

typedef __attribute__((ext_vector_type(16))) __bf16 bf16x16;
typedef __attribute__((ext_vector_type(8)))  float  floatx8;

__device__ __forceinline__ unsigned short f32_to_bf16(float f) {
    union { float f; unsigned int u; } v; v.f = f;
    unsigned int u = v.u;
    unsigned int r = u + 0x7FFFu + ((u >> 16) & 1u);   // round to nearest even
    return (unsigned short)(r >> 16);
}

// ---------------------------------------------------------------------------
// Kernel 0: one-time weight conversion  W_ref (128,144) f32 -> (128,160) bf16
// ---------------------------------------------------------------------------
__global__ __launch_bounds__(256)
void seesaw_convert_wref(const float* __restrict__ Wref,
                         unsigned short* __restrict__ wbf) {
    const int i = blockIdx.x * 256 + threadIdx.x;      // 128*160 = 20480
    if (i < OC_ * KP_) {
        const int n = i / KP_;
        const int k = i - n * KP_;
        wbf[i] = (k < KR_) ? f32_to_bf16(Wref[n * KR_ + k]) : (unsigned short)0;
    }
}

// ---------------------------------------------------------------------------
// Kernel 1: offset conv as implicit GEMM on v_wmma_f32_16x16x32_bf16.
//   Tile: 64 pixels (one h row) x 128 out channels per workgroup.
//   256 threads = 8 wave32; wave w owns N-tile [16w,16w+16), loops 4 M-tiles.
//   B (bf16, preconverted) staged to LDS with global_load_async_to_lds_b128.
//   Writes off in (B, H, W, 128) channel-last layout, bias fused.
// ---------------------------------------------------------------------------
__global__ __launch_bounds__(256)
void seesaw_offset_conv_wmma(const float* __restrict__ x,
                             const unsigned short* __restrict__ wbf,
                             const float* __restrict__ bref,
                             float* __restrict__ off) {
    __shared__ __align__(16) unsigned short Alds[MT_ * KP_];   // 20 KB
    __shared__ __align__(16) unsigned short Blds[OC_ * KP_];   // 40 KB

    const int tile = blockIdx.x;                 // 4*192*3 = 2304 tiles
    const int wt   = tile % (W_ / MT_);
    const int h    = (tile / (W_ / MT_)) % H_;
    const int b    = tile / ((W_ / MT_) * H_);
    const int wbase = wt * MT_;
    const int tid  = threadIdx.x;

    // ---- stage B: 40960 bytes via async global->LDS (ASYNCcnt), 10 x b128/lane ----
    {
        const unsigned lds_b = (unsigned)(unsigned long long)(uintptr_t)&Blds[0];
#pragma unroll
        for (int it = 0; it < 10; ++it) {
            const unsigned byteoff = (unsigned)(it * 256 + tid) * 16u;  // 16B per lane
            const unsigned ldsaddr = lds_b + byteoff;
            const unsigned long long gaddr =
                (unsigned long long)(uintptr_t)wbf + byteoff;
            asm volatile("global_load_async_to_lds_b128 %0, %1, off"
                         :: "v"(ldsaddr), "v"(gaddr) : "memory");
        }
    }

    // ---- stage A: 64x160 bf16 im2col patch (zero pad K in [144,160)) ----
    for (int i = tid; i < MT_ * KP_; i += 256) {
        const int m = i / KP_;          // pixel within tile
        const int k = i - m * KP_;      // reduction index
        float v = 0.0f;
        if (k < KR_) {
            const int ci  = k / 9;
            const int tap = k - ci * 9;
            const int ky  = tap / 3;
            const int kx  = tap - ky * 3;
            const int hh  = h + ky - 1;
            const int ww  = wbase + m + kx - 1;
            if (hh >= 0 && hh < H_ && ww >= 0 && ww < W_)
                v = x[((b * C_ + ci) * H_ + hh) * W_ + ww];
        }
        Alds[m * KP_ + k] = f32_to_bf16(v);
    }

    asm volatile("s_wait_asynccnt 0" ::: "memory");   // async B copies landed
    __syncthreads();                                  // A + B visible to all waves

    const int lane  = tid & 31;
    const int wave  = tid >> 5;
    const int nbase = wave * 16;
    const int nloc  = lane & 15;
    const int khalf = lane >> 4;

    floatx8 acc0 = {}, acc1 = {}, acc2 = {}, acc3 = {};
    const unsigned int* __restrict__ B32 =
        (const unsigned int*)&Blds[(nbase + nloc) * KP_];

#pragma unroll
    for (int kk = 0; kk < KP_; kk += 32) {
        // 16-bit B 32x16, contiguous k order used here: lanes0-15 K=0..15 over V0..7, lanes16-31 K=16..31
        union { unsigned int u[8]; bf16x16 v; } fb;
#pragma unroll
        for (int j = 0; j < 8; ++j)
            fb.u[j] = B32[(kk + 2 * j + khalf * 16) >> 1];

#pragma unroll
        for (int mt = 0; mt < 4; ++mt) {
            // 16-bit A 16x32 layout (matches B): lanes0-15 K=0..15 over V0..7,
            // lanes16-31 K=16..31
            const unsigned int* __restrict__ A32 =
                (const unsigned int*)&Alds[(mt * 16 + nloc) * KP_];
            union { unsigned int u[8]; bf16x16 v; } fa;
#pragma unroll
            for (int j = 0; j < 8; ++j)
                fa.u[j] = A32[(kk + 2 * j + khalf * 16) >> 1];

            floatx8 c = (mt == 0) ? acc0 : (mt == 1) ? acc1 : (mt == 2) ? acc2 : acc3;
            c = __builtin_amdgcn_wmma_f32_16x16x32_bf16(
                /*neg_a=*/false, fa.v, /*neg_b=*/false, fb.v,
                /*c_mod=*/(short)0, c, /*reuse_a=*/false, /*reuse_b=*/false);
            if (mt == 0) acc0 = c; else if (mt == 1) acc1 = c;
            else if (mt == 2) acc2 = c; else acc3 = c;
        }
    }

    // ---- store D: lane/VGPR r -> (m = r + 8*khalf, n = nbase + nloc) ----
    const int nglob = nbase + nloc;
    const float bias = bref[nglob];
#pragma unroll
    for (int mt = 0; mt < 4; ++mt) {
        const floatx8& c = (mt == 0) ? acc0 : (mt == 1) ? acc1 : (mt == 2) ? acc2 : acc3;
#pragma unroll
        for (int r = 0; r < 8; ++r) {
            const int m = mt * 16 + r + 8 * khalf;
            off[((b * H_ + h) * W_ + (wbase + m)) * OC_ + nglob] = c[r] + bias;
        }
    }
}

// ---------------------------------------------------------------------------
// Kernel 2: fused symmetric-offset grid_sample + raw-view scramble +
//           grouped 1x1 conv + residual. One thread per output element.
// ---------------------------------------------------------------------------
__global__ __launch_bounds__(256)
void seesaw_sample_ws(const float* __restrict__ x,
                      const float* __restrict__ Wws,   // (C, KE)
                      const float* __restrict__ bws,   // (C,)
                      const float* __restrict__ off,   // (B,H,W,128)
                      float* __restrict__ out) {       // (B,C,H,W)
    const unsigned int idx = blockIdx.x * 256u + threadIdx.x;   // < 2359296
    const int w = idx % W_;
    const int h = (idx / W_) % H_;
    const int c = (idx / HW_) % C_;
    const int b = idx / (C_ * HW_);

    const float inv191 = 1.0f / 191.0f;
    float acc = 0.0f;

#pragma unroll
    for (int ke = 0; ke < KE_; ++ke) {
        // flat index of scr[b,h,w,c,ke], reinterpreted in (C,KE,B,H,W) order
        unsigned int L = ((((unsigned)b * H_ + h) * W_ + w) * C_ + c) * KE_ + ke;
        const unsigned int c2  = L / KEBHW_;  L -= c2  * KEBHW_;
        const unsigned int ke2 = L / BHW_;    L -= ke2 * BHW_;
        const unsigned int b2  = L / HW_;     L -= b2  * HW_;
        const unsigned int h2  = L / W_;
        const unsigned int w2  = L - h2 * W_;

        // offset pair (component0 -> x/width, component1 -> y/height)
        const unsigned int ob = ((b2 * H_ + h2) * W_ + w2) * OC_ + c2 * 8;
        float ox, oy;
        if (ke2 < 4u) {
            ox = off[ob + 2u * ke2];
            oy = off[ob + 2u * ke2 + 1u];
        } else if (ke2 == 4u) {
            ox = 0.0f; oy = 0.0f;
        } else {
            const unsigned int m2 = ke2 - 5u;
            ox = 2.0f - off[ob + 2u * m2];
            oy = 2.0f - off[ob + 2u * m2 + 1u];
        }

        // base grid: linspace over h into comp0, over w into comp1 (faithful quirk)
        const float xn = ox + (float)h2 * inv191;
        const float yn = oy + (float)w2 * inv191;
        const float ix = ((xn + 1.0f) * (float)W_ - 1.0f) * 0.5f;
        const float iy = ((yn + 1.0f) * (float)H_ - 1.0f) * 0.5f;
        const float x0 = floorf(ix);
        const float y0 = floorf(iy);
        const float wx = ix - x0;
        const float wy = iy - y0;
        const int x0i = (int)x0, y0i = (int)y0;
        const int x1i = x0i + 1, y1i = y0i + 1;

        const float* __restrict__ fplane = x + (b2 * C_ + c2) * HW_;
        __builtin_prefetch(fplane + (y0i >= 0 ? y0i : 0) * W_, 0, 0);

        float v00 = 0.0f, v01 = 0.0f, v10 = 0.0f, v11 = 0.0f;
        const bool yin0 = (y0i >= 0) & (y0i < H_);
        const bool yin1 = (y1i >= 0) & (y1i < H_);
        const bool xin0 = (x0i >= 0) & (x0i < W_);
        const bool xin1 = (x1i >= 0) & (x1i < W_);
        if (yin0 & xin0) v00 = fplane[y0i * W_ + x0i];
        if (yin0 & xin1) v01 = fplane[y0i * W_ + x1i];
        if (yin1 & xin0) v10 = fplane[y1i * W_ + x0i];
        if (yin1 & xin1) v11 = fplane[y1i * W_ + x1i];

        const float val = v00 * (1.0f - wy) * (1.0f - wx)
                        + v01 * (1.0f - wy) * wx
                        + v10 * wy * (1.0f - wx)
                        + v11 * wy * wx;
        acc += val * Wws[c * KE_ + ke];
    }

    out[idx] = acc + bws[c] + x[idx];
}

// ---------------------------------------------------------------------------
// Launch: inputs in setup_inputs() order: x, W_ref, b_ref, W_ws, b_ws
// ---------------------------------------------------------------------------
extern "C" void kernel_launch(void* const* d_in, const int* in_sizes, int n_in,
                              void* d_out, int out_size, void* d_ws, size_t ws_size,
                              hipStream_t stream) {
    const float* x    = (const float*)d_in[0];
    const float* Wref = (const float*)d_in[1];
    const float* bref = (const float*)d_in[2];
    const float* Wws  = (const float*)d_in[3];
    const float* bws  = (const float*)d_in[4];
    float* out = (float*)d_out;

    float* off = (float*)d_ws;                               // (B,H,W,128) f32, 75.5 MB
    unsigned short* wbf =
        (unsigned short*)(off + (size_t)BS * HW_ * OC_);     // (128,160) bf16, 40 KB

    seesaw_convert_wref<<<(OC_ * KP_ + 255) / 256, 256, 0, stream>>>(Wref, wbf);

    const int tiles = BS * H_ * (W_ / MT_);                  // 2304
    seesaw_offset_conv_wmma<<<tiles, 256, 0, stream>>>(x, wbf, bref, off);

    const int total = BS * C_ * HW_;                         // 2359296
    seesaw_sample_ws<<<total / 256, 256, 0, stream>>>(x, Wws, bws, off, out);
}
